// BiGAT_67662914781638
// MI455X (gfx1250) — matrix operand order repaired and checked by reading, not compiled, not run on hardware
//
#include <hip/hip_runtime.h>
#include <hip/hip_bf16.h>

// ---------------------------------------------------------------------------
// BiGAT (2x GATConv per branch, heads=1) for MI455X / gfx1250.
//  - GEMMs via v_wmma_f32_16x16x32_bf16 (bf16 inputs, f32 accumulate)
//  - CSR-based attention aggregation (no atomics on the 128-wide gather path)
//  - global_prefetch on gathered feature rows, LDS block scans, f32 atomics
// ---------------------------------------------------------------------------

typedef __attribute__((ext_vector_type(16))) __bf16 v16bf;
typedef __attribute__((ext_vector_type(8)))  __bf16 v8bf;
typedef __attribute__((ext_vector_type(4)))  __bf16 v4bf;
typedef __attribute__((ext_vector_type(8)))  float  v8f;

#define FDIM 128
#define NEG_SLOPE 0.2f

static __device__ __forceinline__ float wave_sum(float v) {
#pragma unroll
  for (int m = 16; m >= 1; m >>= 1) v += __shfl_xor(v, m, 32);
  return v;
}
static __device__ __forceinline__ float wave_max(float v) {
#pragma unroll
  for (int m = 16; m >= 1; m >>= 1) v = fmaxf(v, __shfl_xor(v, m, 32));
  return v;
}
static __device__ __forceinline__ float lrelu(float e) {
  return e > 0.f ? e : NEG_SLOPE * e;
}

// ------------------------------- utilities ---------------------------------
__global__ void zerof_k(float* p, int n) {
  int i = blockIdx.x * 256 + threadIdx.x;
  if (i < n) p[i] = 0.f;
}
__global__ void zeroi_k(int* p, int n) {
  int i = blockIdx.x * 256 + threadIdx.x;
  if (i < n) p[i] = 0;
}
__global__ void cvt_k(const float* __restrict__ x, __bf16* __restrict__ xb,
                      int n_valid, int n_total) {
  int i = blockIdx.x * 256 + threadIdx.x;
  if (i >= n_total) return;
  float v = (i < n_valid) ? x[i] : 0.f;
  xb[i] = (__bf16)v;
}
__global__ void batch_count_k(const int* __restrict__ batch, float* gcnt, int n) {
  int i = blockIdx.x * 256 + threadIdx.x;
  if (i < n) atomicAdd(&gcnt[batch[i]], 1.f);
}

// ----------------------------- CSR construction ----------------------------
__global__ void count_k(const int* __restrict__ ei, int* cnt, int E, int n) {
  int i = blockIdx.x * 256 + threadIdx.x;
  int en = E + n;
  if (i >= en) return;
  int d = (i < E) ? ei[E + i] : (i - E);  // self loops appended
  atomicAdd(&cnt[d], 1);
}

// block scan over 1024 elements (256 threads x 4)
__global__ __launch_bounds__(256) void scan_blocks_k(const int* __restrict__ cnt,
                                                     int* __restrict__ rowstart,
                                                     int* __restrict__ bsums, int n) {
  __shared__ int sh[256];
  int t = threadIdx.x;
  int base = blockIdx.x * 1024;
  int v[4];
  int s = 0;
#pragma unroll
  for (int e = 0; e < 4; ++e) {
    int i = base + t * 4 + e;
    v[e] = (i < n) ? cnt[i] : 0;
    s += v[e];
  }
  sh[t] = s;
  __syncthreads();
  for (int off = 1; off < 256; off <<= 1) {
    int x = (t >= off) ? sh[t - off] : 0;
    __syncthreads();
    sh[t] += x;
    __syncthreads();
  }
  int run = sh[t] - s;  // block-local exclusive
#pragma unroll
  for (int e = 0; e < 4; ++e) {
    int i = base + t * 4 + e;
    if (i < n) rowstart[i] = run;
    run += v[e];
  }
  if (t == 255) bsums[blockIdx.x] = sh[t];
}

// single-block scan of block sums (nb <= 256; N=100k -> nb=98)
__global__ __launch_bounds__(256) void scan_top_k(int* bsums, int nb) {
  __shared__ int sh[256];
  int t = threadIdx.x;
  int v = (t < nb) ? bsums[t] : 0;
  sh[t] = v;
  __syncthreads();
  for (int off = 1; off < 256; off <<= 1) {
    int x = (t >= off) ? sh[t - off] : 0;
    __syncthreads();
    sh[t] += x;
    __syncthreads();
  }
  if (t < nb) bsums[t] = sh[t] - v;  // exclusive
}

__global__ void scan_fix_k(int* rowstart, const int* __restrict__ bsums, int* cnt,
                           int n, int total) {
  int i = blockIdx.x * 256 + threadIdx.x;
  if (i < n) {
    rowstart[i] += bsums[i >> 10];
    cnt[i] = 0;  // reuse as scatter cursor
  }
  if (i == 0) rowstart[n] = total;
}

__global__ void scatter_k(const int* __restrict__ ei, const int* __restrict__ rowstart,
                          int* cur, int* __restrict__ esrc, int E, int n) {
  int i = blockIdx.x * 256 + threadIdx.x;
  int en = E + n;
  if (i >= en) return;
  int s, d;
  if (i < E) { s = ei[i]; d = ei[E + i]; } else { s = i - E; d = s; }
  int pos = rowstart[d] + atomicAdd(&cur[d], 1);
  esrc[pos] = s;
}

// ------------------------------ weight packing -----------------------------
// Wp element index = ((ct*4 + kc)*32 + lane)*16 + e
// B fragment (32x16 bf16, 16x16x32 wmma): col = ct*16 + lane%16,
// K = kc*32 + (lane<16 ? 0 : 8) + (e<8 ? e : 8+e)       (W is row-major [K=128, N=128])
__global__ void packW_k(const float* __restrict__ W, __bf16* __restrict__ Wp) {
  int idx = blockIdx.x * 256 + threadIdx.x;  // 16384 total
  int e    = idx & 15;
  int lane = (idx >> 4) & 31;
  int kc   = (idx >> 9) & 3;
  int ct   = idx >> 11;
  int col  = ct * 16 + (lane & 15);
  int k    = kc * 32 + ((lane < 16) ? 0 : 8) + ((e < 8) ? e : 8 + e);
  Wp[idx] = (__bf16)W[k * FDIM + col];
}

// --------------------------------- GEMM ------------------------------------
// C[NPAD,128] = A[NPAD,128](bf16) @ W(packed bf16), f32 accumulate.
// Block: 256 thr = 8 waves; block tile 64 rows x 128 cols;
// wave tile 16 rows x 64 cols = 4 wmma tiles, K-loop 4 x 32.
__global__ __launch_bounds__(256) void gemm_k(const __bf16* __restrict__ A,
                                              const __bf16* __restrict__ Wp,
                                              float* __restrict__ C, int n) {
  int wave = threadIdx.x >> 5;
  int lane = threadIdx.x & 31;
  int rowTile = wave >> 1;   // 0..3
  int colHalf = wave & 1;    // 0..1
  int r0 = blockIdx.x * 64 + rowTile * 16;

  // A fragment addressing (ISA 16-bit A 16x32 layout)
  int arow = r0 + (lane & 15);
  int koff = (lane < 16) ? 0 : 8;

  v8f c[4] = {v8f{}, v8f{}, v8f{}, v8f{}};
#pragma unroll
  for (int kc = 0; kc < 4; ++kc) {
    const __bf16* ap = A + arow * FDIM + kc * 32 + koff;
    v8bf alo = *(const v8bf*)(ap);        // elements 0..7
    v8bf ahi = *(const v8bf*)(ap + 16);   // elements 8..15
    v16bf a = __builtin_shufflevector(alo, ahi, 0, 1, 2, 3, 4, 5, 6, 7,
                                      8, 9, 10, 11, 12, 13, 14, 15);
#pragma unroll
    for (int t = 0; t < 4; ++t) {
      int ct = colHalf * 4 + t;
      const v16bf b = *(const v16bf*)(Wp + ((ct * 4 + kc) * 32 + lane) * 16);
      c[t] = __builtin_amdgcn_wmma_f32_16x16x32_bf16(
          /*neg_a=*/false, a, /*neg_b=*/false, b,
          /*c_mod=*/(short)0, c[t], /*reuse_a=*/false, /*reuse_b=*/false);
    }
  }
  // D layout: VGPR r, lanes 0-15 -> (M=r, N=lane), lanes 16-31 -> (M=r+8)
  int colb  = colHalf * 64 + (lane & 15);
  int rbase = r0 + ((lane < 16) ? 0 : 8);
#pragma unroll
  for (int r = 0; r < 8; ++r) {
    int row = rbase + r;
    if (row < n) {
      float* cp = C + row * FDIM + colb;
      cp[0]  = c[0][r];
      cp[16] = c[1][r];
      cp[32] = c[2][r];
      cp[48] = c[3][r];
    }
  }
}

// ------------------------- attention logits per node -----------------------
__global__ __launch_bounds__(256) void alphas_k(const float* __restrict__ hf,
                                                const float* __restrict__ as,
                                                const float* __restrict__ ad,
                                                float* als, float* ald, int n) {
  int lane = threadIdx.x & 31;
  int row = blockIdx.x * 8 + (threadIdx.x >> 5);
  if (row >= n) return;
  const float4 hv = *(const float4*)(hf + row * FDIM + lane * 4);
  const float4 sa = *(const float4*)(as + lane * 4);
  const float4 da = *(const float4*)(ad + lane * 4);
  float ps = hv.x * sa.x + hv.y * sa.y + hv.z * sa.z + hv.w * sa.w;
  float pd = hv.x * da.x + hv.y * da.y + hv.z * da.z + hv.w * da.w;
  ps = wave_sum(ps);
  pd = wave_sum(pd);
  if (lane == 0) { als[row] = ps; ald[row] = pd; }
}

// ------------------- fused softmax + gather aggregation --------------------
// One wave per destination node. mode 0: relu(out+b) -> bf16 activations.
// mode 1: relu(out+b) -> atomic segment-sum into per-graph accumulator.
__global__ __launch_bounds__(256) void aggregate_k(
    const float* __restrict__ hf, const float* __restrict__ als,
    const float* __restrict__ ald, const int* __restrict__ rowstart,
    const int* __restrict__ esrc, const float* __restrict__ bias,
    __bf16* __restrict__ outb, float* __restrict__ gsum,
    const int* __restrict__ batch, int n, int mode) {
  int lane = threadIdx.x & 31;
  int row = blockIdx.x * 8 + (threadIdx.x >> 5);
  if (row >= n) return;

  int d0 = rowstart[row], d1 = rowstart[row + 1];
  float adi = ald[row];

  // pass 1: max over incoming edges (self loop guarantees d1 > d0)
  float mx = -3.4e38f;
  for (int j = d0 + lane; j < d1; j += 32)
    mx = fmaxf(mx, lrelu(als[esrc[j]] + adi));
  mx = wave_max(mx);

  // pass 2: exp-sum
  float sm = 0.f;
  for (int j = d0 + lane; j < d1; j += 32)
    sm += __expf(lrelu(als[esrc[j]] + adi) - mx);
  sm = wave_sum(sm);
  float rs = 1.f / sm;

  // pass 3: weighted gather of h[src]; 4 features per lane
  int colb = lane * 4;
  float a0 = 0.f, a1 = 0.f, a2 = 0.f, a3 = 0.f;
  for (int j = d0; j < d1; ++j) {
    int s = esrc[j];
    if (j + 1 < d1)
      __builtin_prefetch((const void*)(hf + esrc[j + 1] * FDIM + colb), 0, 1);
    float coef = __expf(lrelu(als[s] + adi) - mx) * rs;
    const float4 hv = *(const float4*)(hf + s * FDIM + colb);
    a0 += coef * hv.x; a1 += coef * hv.y; a2 += coef * hv.z; a3 += coef * hv.w;
  }
  const float4 bv = *(const float4*)(bias + colb);
  float o0 = fmaxf(a0 + bv.x, 0.f);
  float o1 = fmaxf(a1 + bv.y, 0.f);
  float o2 = fmaxf(a2 + bv.z, 0.f);
  float o3 = fmaxf(a3 + bv.w, 0.f);

  if (mode == 0) {
    v4bf pv = { (__bf16)o0, (__bf16)o1, (__bf16)o2, (__bf16)o3 };
    *(v4bf*)(outb + row * FDIM + colb) = pv;
  } else {
    float* gp = gsum + batch[row] * FDIM + colb;
    atomicAdd(gp + 0, o0);
    atomicAdd(gp + 1, o1);
    atomicAdd(gp + 2, o2);
    atomicAdd(gp + 3, o3);
  }
}

// ------------------------- readout: fc + log_softmax -----------------------
__global__ __launch_bounds__(256) void final_k(const float* __restrict__ gsum_bu,
                                               const float* __restrict__ gsum_td,
                                               const float* __restrict__ gcnt,
                                               const float* __restrict__ fcW,
                                               const float* __restrict__ fcb,
                                               float* __restrict__ out, int G) {
  int lane = threadIdx.x & 31;
  int g = blockIdx.x * 8 + (threadIdx.x >> 5);
  if (g >= G) return;
  float inv = 1.f / fmaxf(gcnt[g], 1.f);
  float z0 = 0.f, z1 = 0.f, z2 = 0.f, z3 = 0.f;
#pragma unroll
  for (int e = 0; e < 8; ++e) {
    int f = lane * 8 + e;  // concat order: [bu(0..127), td(128..255)]
    float v = ((f < FDIM) ? gsum_bu[g * FDIM + f] : gsum_td[g * FDIM + f - FDIM]) * inv;
    const float4 w = *(const float4*)(fcW + f * 4);
    z0 += v * w.x; z1 += v * w.y; z2 += v * w.z; z3 += v * w.w;
  }
  z0 = wave_sum(z0); z1 = wave_sum(z1); z2 = wave_sum(z2); z3 = wave_sum(z3);
  if (lane == 0) {
    z0 += fcb[0]; z1 += fcb[1]; z2 += fcb[2]; z3 += fcb[3];
    float mz = fmaxf(fmaxf(z0, z1), fmaxf(z2, z3));
    float lse = mz + logf(__expf(z0 - mz) + __expf(z1 - mz) +
                          __expf(z2 - mz) + __expf(z3 - mz));
    out[g * 4 + 0] = z0 - lse;
    out[g * 4 + 1] = z1 - lse;
    out[g * 4 + 2] = z2 - lse;
    out[g * 4 + 3] = z3 - lse;
  }
}

// ---------------------------------------------------------------------------
extern "C" void kernel_launch(void* const* d_in, const int* in_sizes, int n_in,
                              void* d_out, int out_size, void* d_ws, size_t ws_size,
                              hipStream_t stream) {
  const int F = FDIM;
  const int N = in_sizes[0] / F;        // 100000
  const int E = in_sizes[1] / 2;        // 1600000
  const int EN = E + N;                 // edges + self loops
  const int G = out_size / 4;           // 512
  const int NPAD = ((N + 63) / 64) * 64;

  const float* x     = (const float*)d_in[0];
  const int*   td_ei = (const int*)d_in[1];
  const int*   bu_ei = (const int*)d_in[2];
  const int*   batch = (const int*)d_in[3];
  const float* fcW   = (const float*)d_in[20];
  const float* fcb   = (const float*)d_in[21];

  // ---- workspace carving (all buffers fully re-initialized per call) ----
  char* p = (char*)d_ws;
  auto carve = [&](size_t bytes) -> char* {
    char* r = p;
    p += (bytes + 255) & ~(size_t)255;
    return r;
  };
  __bf16* xb      = (__bf16*)carve((size_t)NPAD * F * 2);
  __bf16* hb      = (__bf16*)carve((size_t)NPAD * F * 2);
  float*  hf      = (float*) carve((size_t)NPAD * F * 4);
  __bf16* Wp      = (__bf16*)carve((size_t)F * F * 2);
  float*  als     = (float*) carve((size_t)N * 4);
  float*  ald     = (float*) carve((size_t)N * 4);
  int*    rowstart= (int*)   carve((size_t)(N + 1) * 4);
  int*    cnt     = (int*)   carve((size_t)N * 4);
  int*    esrc    = (int*)   carve((size_t)EN * 4);
  int*    bsums   = (int*)   carve(4096);
  float*  gsum_td = (float*) carve((size_t)G * F * 4);
  float*  gsum_bu = (float*) carve((size_t)G * F * 4);
  float*  gcnt    = (float*) carve((size_t)G * 4);

  auto cdiv = [](int a, int b) { return (a + b - 1) / b; };
  const int nb = cdiv(N, 1024);  // scan blocks (98 for N=100k, must be <=256)

  zerof_k<<<cdiv(G * F, 256), 256, 0, stream>>>(gsum_td, G * F);
  zerof_k<<<cdiv(G * F, 256), 256, 0, stream>>>(gsum_bu, G * F);
  zerof_k<<<cdiv(G, 256), 256, 0, stream>>>(gcnt, G);
  batch_count_k<<<cdiv(N, 256), 256, 0, stream>>>(batch, gcnt, N);
  cvt_k<<<cdiv(NPAD * F, 256), 256, 0, stream>>>(x, xb, N * F, NPAD * F);

  for (int br = 0; br < 2; ++br) {
    const int* ei = (br == 0) ? td_ei : bu_ei;
    const int base = (br == 0) ? 4 : 12;
    const float* W1  = (const float*)d_in[base + 0];
    const float* as1 = (const float*)d_in[base + 1];
    const float* ad1 = (const float*)d_in[base + 2];
    const float* b1  = (const float*)d_in[base + 3];
    const float* W2  = (const float*)d_in[base + 4];
    const float* as2 = (const float*)d_in[base + 5];
    const float* ad2 = (const float*)d_in[base + 6];
    const float* b2  = (const float*)d_in[base + 7];
    float* gs = (br == 0) ? gsum_td : gsum_bu;

    // CSR build (incoming-edge lists incl. self loops)
    zeroi_k<<<cdiv(N, 256), 256, 0, stream>>>(cnt, N);
    count_k<<<cdiv(EN, 256), 256, 0, stream>>>(ei, cnt, E, N);
    scan_blocks_k<<<nb, 256, 0, stream>>>(cnt, rowstart, bsums, N);
    scan_top_k<<<1, 256, 0, stream>>>(bsums, nb);
    scan_fix_k<<<cdiv(N, 256), 256, 0, stream>>>(rowstart, bsums, cnt, N, EN);
    scatter_k<<<cdiv(EN, 256), 256, 0, stream>>>(ei, rowstart, cnt, esrc, E, N);

    // layer 1: h1 = x @ W1 -> attention -> relu -> bf16 activations
    packW_k<<<64, 256, 0, stream>>>(W1, Wp);
    gemm_k<<<NPAD / 64, 256, 0, stream>>>(xb, Wp, hf, N);
    alphas_k<<<cdiv(N, 8), 256, 0, stream>>>(hf, as1, ad1, als, ald, N);
    aggregate_k<<<cdiv(N, 8), 256, 0, stream>>>(hf, als, ald, rowstart, esrc, b1,
                                                hb, nullptr, nullptr, N, 0);
    // layer 2: h2 = h1 @ W2 -> attention -> relu -> per-graph segment sum
    packW_k<<<64, 256, 0, stream>>>(W2, Wp);
    gemm_k<<<NPAD / 64, 256, 0, stream>>>(hb, Wp, hf, N);
    alphas_k<<<cdiv(N, 8), 256, 0, stream>>>(hf, as2, ad2, als, ald, N);
    aggregate_k<<<cdiv(N, 8), 256, 0, stream>>>(hf, als, ald, rowstart, esrc, b2,
                                                nullptr, gs, batch, N, 1);
  }

  final_k<<<cdiv(G, 8), 256, 0, stream>>>(gsum_bu, gsum_td, gcnt, fcW, fcb,
                                          (float*)d_out, G);
}
